// Attention_45681272160684
// MI455X (gfx1250) — matrix operand order repaired and checked by reading, not compiled, not run on hardware
//
#include <hip/hip_runtime.h>
#include <hip/hip_bf16.h>

// ---- problem constants (from reference) ----
#define BATCH     2
#define SEQ       2048
#define CDIM      2048
#define NHEADS    16
#define NKVHEADS  4
#define HDIM      128
#define QKV_N     3072   // 2048 (q) + 512 (k) + 512 (v)

typedef __attribute__((ext_vector_type(16))) __bf16      v16bf;
typedef __attribute__((ext_vector_type(8)))  float       v8f;
typedef __attribute__((ext_vector_type(4))) unsigned     u32x4;
typedef __attribute__((ext_vector_type(2))) unsigned     u32x2;
typedef __attribute__((ext_vector_type(4)))  float       f32x4;
typedef __attribute__((ext_vector_type(8)))  int         i32x8;
typedef __attribute__((ext_vector_type(4)))  int         i32x4;

__device__ inline v8f v8f_zero() {
    v8f z = {0.f, 0.f, 0.f, 0.f, 0.f, 0.f, 0.f, 0.f};
    return z;
}

__device__ inline unsigned short f32_to_bf16(float f) {
    union { float f; unsigned u; } a; a.f = f;
    unsigned u = a.u;
    u += 0x7FFFu + ((u >> 16) & 1u);   // round-to-nearest-even
    return (unsigned short)(u >> 16);
}

__device__ inline unsigned pack_bf16x2(float lo, float hi) {
    return (unsigned)f32_to_bf16(lo) | ((unsigned)f32_to_bf16(hi) << 16);
}

// Low 32 bits of a generic pointer to __shared__ = LDS byte address
// (ISA 10.2: aperture LDS translation keeps addr[31:0]).
__device__ inline unsigned lds_u32(const void* p) {
    return (unsigned)(unsigned long long)(size_t)p;
}

// Async global->LDS copy, 16B per active lane. Tracked by ASYNCcnt.
__device__ inline void async_load_b128(unsigned lds_off, const void* gaddr) {
    asm volatile("global_load_async_to_lds_b128 %0, %1, off"
                 :: "v"(lds_off), "v"(gaddr) : "memory");
}
__device__ inline void wait_asynccnt0() {
    asm volatile("s_wait_asynccnt 0x0" ::: "memory");
}

// ---------------------------------------------------------------------------
// Tensor Data Mover: 2D tile load (bf16, data_size=2B) global -> LDS.
// D# bit layout per CDNA5 ISA §8 (group0: count/lds/global/type,
// group1: data_size, tensor dims, tile dims, dim0 stride).
// This toolchain exposes the 6-arg builtin (extra i32x8 descriptor group).
// ---------------------------------------------------------------------------
__device__ inline void tdm_load_2d_bf16(unsigned lds_addr, const void* gp,
                                        unsigned tile0, unsigned tile1,
                                        unsigned long long stride0,
                                        unsigned dim0, unsigned dim1) {
    unsigned long long ga = (unsigned long long)(size_t)gp;
    u32x4 g0;
    g0[0] = 1u;                                         // count=1, user mode
    g0[1] = lds_addr;                                   // lds_addr [63:32]
    g0[2] = (unsigned)ga;                               // global_addr [95:64]
    g0[3] = (unsigned)((ga >> 32) & 0x01FFFFFFu)        // global_addr [120:96]
            | (2u << 30);                               // type=2 ("image")
    i32x8 g1;
    g1[0] = (int)(1u << 16);                            // data_size=1 (2B), mask=0
    g1[1] = (int)((dim0 & 0xFFFFu) << 16);              // tensor_dim0[15:0]
    g1[2] = (int)(((dim0 >> 16) & 0xFFFFu) | ((dim1 & 0xFFFFu) << 16));
    g1[3] = (int)(((dim1 >> 16) & 0xFFFFu) | (tile0 << 16));   // tile_dim0
    g1[4] = (int)(tile1 & 0xFFFFu);                     // tile_dim1 (tile_dim2=0)
    g1[5] = (int)(unsigned)(stride0 & 0xFFFFFFFFull);   // tensor_dim0_stride lo
    g1[6] = (int)(unsigned)((stride0 >> 32) & 0xFFFFull);
    g1[7] = 0;
    i32x4 g2 = {0, 0, 0, 0};
    i32x4 g3 = {0, 0, 0, 0};
    i32x8 g4 = {0, 0, 0, 0, 0, 0, 0, 0};
    __builtin_amdgcn_tensor_load_to_lds(g0, g1, g2, g3, g4, 0);
}

union FragU { v16bf v; u32x4 u[2]; unsigned short s[16]; };

// A fragment (16x32 bf16, M across lanes, K packed):
// lanes 0-15 : M = row+lane,    K = {0..7, 16..23}
// lanes 16-31: M = row+lane-16, K = {8..15, 24..31}
__device__ inline v16bf frag_a(const unsigned short* base, int row, int stride, int lane) {
    int r  = row + (lane & 15);
    int ko = (lane & 16) ? 8 : 0;
    FragU f;
    f.u[0] = *(const u32x4*)(base + (size_t)r * stride + ko);
    f.u[1] = *(const u32x4*)(base + (size_t)r * stride + ko + 16);
    return f.v;
}

// B fragment (32x16 bf16, N across lanes, K packed) from N-major storage:
// lanes 0-15 : N = col+lane,    K = {0..7, 16..23}
// lanes 16-31: N = col+lane-16, K = {8..15, 24..31}
__device__ inline v16bf frag_b(const unsigned short* base, int col, int stride, int lane) {
    int n  = col + (lane & 15);
    int ko = (lane & 16) ? 8 : 0;
    FragU f;
    f.u[0] = *(const u32x4*)(base + (size_t)n * stride + ko);
    f.u[1] = *(const u32x4*)(base + (size_t)n * stride + ko + 16);
    return f.v;
}

// A fragment assembled from f32 data (probabilities), converting to bf16.
__device__ inline v16bf frag_a_f32(const float* base, int row, int stride, int lane) {
    int r  = row + (lane & 15);
    int ko = (lane & 16) ? 8 : 0;
    const float* p = base + (size_t)r * stride + ko;
    FragU f;
#pragma unroll
    for (int i = 0; i < 8; ++i) {
        f.s[i]     = f32_to_bf16(p[i]);
        f.s[i + 8] = f32_to_bf16(p[i + 16]);
    }
    return f.v;
}

__device__ inline v8f wmma_bf16(v16bf a, v16bf b, v8f c) {
    return __builtin_amdgcn_wmma_f32_16x16x32_bf16(false, a, false, b, (short)0, c, false, false);
}

// ---------------------------------------------------------------------------
// Pre-pass: f32 -> bf16 straight conversion (n multiple of 1024).
// ---------------------------------------------------------------------------
__global__ void __launch_bounds__(256)
cvt_f32_bf16(const float* __restrict__ s, unsigned short* __restrict__ d, int n) {
    int i = (blockIdx.x * 256 + threadIdx.x) * 4;
    if (i < n) {
        f32x4 v = *(const f32x4*)(s + i);
        u32x2 h; h[0] = pack_bf16x2(v[0], v[1]); h[1] = pack_bf16x2(v[2], v[3]);
        *(u32x2*)(d + i) = h;
    }
}

// ---------------------------------------------------------------------------
// Pre-pass: f32 [K][N] -> bf16 [N][K] (transpose via LDS tile).
// Grid (N/32, K/32), 256 threads.
// ---------------------------------------------------------------------------
__global__ void __launch_bounds__(256)
cvt_transpose_bf16(const float* __restrict__ s, unsigned short* __restrict__ d,
                   int K, int N) {
    __shared__ unsigned short tile[32][33];
    const int k0 = blockIdx.y * 32, n0 = blockIdx.x * 32;
    const int tx = threadIdx.x & 31, ty = threadIdx.x >> 5;   // ty: 0..7
#pragma unroll
    for (int r = ty; r < 32; r += 8)
        tile[r][tx] = f32_to_bf16(s[(size_t)(k0 + r) * N + n0 + tx]);
    __syncthreads();
#pragma unroll
    for (int r = ty; r < 32; r += 8)
        d[(size_t)(n0 + r) * K + k0 + tx] = tile[tx][r];
}

// ---------------------------------------------------------------------------
// GEMM: C[M x N] = A[M x K] * Bt[N x K]^T, both bf16 row-major-along-K.
// Block tile 128x128, 4 waves of 64x64, K-step 32 (= WMMA depth).
// Tiles staged into LDS by the Tensor Data Mover (one op per tile).
// ---------------------------------------------------------------------------
__global__ void __launch_bounds__(128)
gemm_bf16_wmma(const unsigned short* __restrict__ Ab,
               const unsigned short* __restrict__ Bt,
               float* __restrict__ Cp, int M, int N, int K, int ldc, int col0) {
    __shared__ __align__(16) unsigned short As[128 * 32];
    __shared__ __align__(16) unsigned short Bs[128 * 32];

    const int tid  = threadIdx.x;
    const int lane = tid & 31;
    const int wave = tid >> 5;
    const int m0   = blockIdx.y * 128;
    const int n0   = blockIdx.x * 128;
    const int wm   = (wave >> 1) * 64;
    const int wn   = (wave & 1) * 64;

    v8f acc[4][4];
#pragma unroll
    for (int i = 0; i < 4; ++i)
#pragma unroll
        for (int j = 0; j < 4; ++j) acc[i][j] = v8f_zero();

    const unsigned as_lds = lds_u32(As);
    const unsigned bs_lds = lds_u32(Bs);

    for (int k0 = 0; k0 < K; k0 += 32) {
        if (wave == 0) {   // one wave drives the TDM; EXEC is ignored by TDM
            tdm_load_2d_bf16(as_lds, Ab + (size_t)m0 * K + k0,
                             32, 128, (unsigned long long)K, (unsigned)K, (unsigned)M);
            tdm_load_2d_bf16(bs_lds, Bt + (size_t)n0 * K + k0,
                             32, 128, (unsigned long long)K, (unsigned)K, (unsigned)N);
            __builtin_amdgcn_s_wait_tensorcnt(0);
        }
        __syncthreads();

        v16bf af[4], bfv[4];
#pragma unroll
        for (int mt = 0; mt < 4; ++mt) af[mt]  = frag_a(As, wm + mt * 16, 32, lane);
#pragma unroll
        for (int nt = 0; nt < 4; ++nt) bfv[nt] = frag_b(Bs, wn + nt * 16, 32, lane);
#pragma unroll
        for (int mt = 0; mt < 4; ++mt)
#pragma unroll
            for (int nt = 0; nt < 4; ++nt)
                acc[mt][nt] = wmma_bf16(af[mt], bfv[nt], acc[mt][nt]);
        __syncthreads();
    }

    const int hi8 = (lane & 16) ? 8 : 0;
    const int cl  = lane & 15;
#pragma unroll
    for (int mt = 0; mt < 4; ++mt)
#pragma unroll
        for (int nt = 0; nt < 4; ++nt)
#pragma unroll
            for (int j = 0; j < 8; ++j) {
                int row = m0 + wm + mt * 16 + j + hi8;
                int col = n0 + wn + nt * 16 + cl;
                Cp[(size_t)row * ldc + col0 + col] = acc[mt][nt][j];
            }
}

// ---------------------------------------------------------------------------
// RoPE + repack: proj f32 [B*T, 3072] ->
//   Qb bf16 [B,16,T,128] (RoPE), Kb bf16 [B,4,T,128] (RoPE),
//   Vt bf16 [B,4,128,T]  (transposed, no RoPE).
// Flat head hf = r*4 + h (reference reshape order) -> kv head = hf & 3.
// ---------------------------------------------------------------------------
__global__ void __launch_bounds__(128)
rope_repack(const float* __restrict__ proj, unsigned short* __restrict__ Qb,
            unsigned short* __restrict__ Kb, unsigned short* __restrict__ Vt) {
    const int t = blockIdx.x, b = blockIdx.y;
    const float* row = proj + ((size_t)b * SEQ + t) * QKV_N;
    const int d = threadIdx.x & 63;
    const float inv = __expf(-(float)d * 0.14391156831212787f);  // ln(1e4)/64
    float sn, cs;
    __sincosf((float)t * inv, &sn, &cs);

    for (int it = 0; it < 12; ++it) {
        int slot = it * 2 + (threadIdx.x >> 6);   // 0..23
        if (slot < 16) {                           // Q heads
            float x0 = row[slot * 128 + d], x1 = row[slot * 128 + d + 64];
            size_t base = (((size_t)(b * 16 + slot)) * SEQ + t) * HDIM + d;
            Qb[base]      = f32_to_bf16(x0 * cs - x1 * sn);
            Qb[base + 64] = f32_to_bf16(x1 * cs + x0 * sn);
        } else if (slot < 20) {                    // K heads
            int h = slot - 16;
            float x0 = row[2048 + h * 128 + d], x1 = row[2048 + h * 128 + d + 64];
            size_t base = (((size_t)(b * 4 + h)) * SEQ + t) * HDIM + d;
            Kb[base]      = f32_to_bf16(x0 * cs - x1 * sn);
            Kb[base + 64] = f32_to_bf16(x1 * cs + x0 * sn);
        } else {                                   // V heads, transposed store
            int h = slot - 20;
            float x0 = row[2560 + h * 128 + d], x1 = row[2560 + h * 128 + d + 64];
            size_t base = (((size_t)(b * 4 + h)) * HDIM + d) * (size_t)SEQ + t;
            Vt[base]                    = f32_to_bf16(x0);
            Vt[base + 64 * (size_t)SEQ] = f32_to_bf16(x1);
        }
    }
}

// ---------------------------------------------------------------------------
// Flash attention w/ softcap + causal mask. Grid (32 qblocks, 16 heads, 2 B),
// 128 threads (4 waves); wave w owns query rows w*16..w*16+15.
// K/V tiles staged via async global->LDS loads (ASYNCcnt).
// ---------------------------------------------------------------------------
__global__ void __launch_bounds__(128)
attention_kernel(const unsigned short* __restrict__ Qb,
                 const unsigned short* __restrict__ Kb,
                 const unsigned short* __restrict__ Vt,
                 unsigned short* __restrict__ Ctx) {
    __shared__ __align__(16) unsigned short Ks[64 * 128];   // [key][d]
    __shared__ __align__(16) unsigned short Vs[128 * 64];   // [d][key]
    __shared__ __align__(16) float Sf[64 * 64];
    __shared__ float rowScale[64];
    __shared__ float rowLinv[64];

    const int qb = blockIdx.x, hf = blockIdx.y, b = blockIdx.z;
    const int kvh  = hf & 3;
    const int tid  = threadIdx.x;
    const int lane = tid & 31;
    const int wave = tid >> 5;
    const int hi8  = (lane & 16) ? 8 : 0;
    const int cl   = lane & 15;

    const unsigned short* Qh = Qb + ((size_t)(b * 16 + hf))  * SEQ * HDIM;
    const unsigned short* Kh = Kb + ((size_t)(b * 4 + kvh)) * SEQ * HDIM;
    const unsigned short* Vh = Vt + ((size_t)(b * 4 + kvh)) * HDIM * SEQ;

    const unsigned ks_lds = lds_u32(Ks);
    const unsigned vs_lds = lds_u32(Vs);

    // preload this wave's Q fragments (16 rows x full head dim)
    v16bf qf[4];
#pragma unroll
    for (int ks = 0; ks < 4; ++ks)
        qf[ks] = frag_a(Qh + (size_t)(qb * 64) * HDIM + ks * 32, wave * 16, HDIM, lane);

    v8f accO[8];
#pragma unroll
    for (int vt = 0; vt < 8; ++vt) accO[vt] = v8f_zero();

    float m_i = -1e30f, l_i = 0.f;           // used by tid < 64 only

    for (int kb = 0; kb <= qb; ++kb) {
        // ---- async-stage K tile (64x128, contiguous) and V tile (128x64) ----
        {
            const char* Ksrc = (const char*)(Kh + (size_t)(kb * 64) * HDIM);
#pragma unroll
            for (int i = 0; i < 8; ++i) {
                int chunk = i * 128 + tid;               // 0..1023, 16B each
                async_load_b128(ks_lds + chunk * 16, Ksrc + (size_t)chunk * 16);
            }
#pragma unroll
            for (int i = 0; i < 8; ++i) {
                int chunk = i * 128 + tid;               // row-major [d][8 chunks]
                int rowd = chunk >> 3, c = chunk & 7;
                async_load_b128(vs_lds + chunk * 16,
                                (const char*)(Vh + (size_t)rowd * SEQ + kb * 64) + c * 16);
            }
            wait_asynccnt0();
        }
        __syncthreads();

        // ---- S strip = Q K^T (16 rows x 64 keys) ----
        v8f accS[4];
#pragma unroll
        for (int nt = 0; nt < 4; ++nt) accS[nt] = v8f_zero();
#pragma unroll
        for (int ks = 0; ks < 4; ++ks)
#pragma unroll
            for (int nt = 0; nt < 4; ++nt) {
                v16bf kf = frag_b(Ks + ks * 32, nt * 16, HDIM, lane);
                accS[nt] = wmma_bf16(qf[ks], kf, accS[nt]);
            }
#pragma unroll
        for (int nt = 0; nt < 4; ++nt)
#pragma unroll
            for (int j = 0; j < 8; ++j)
                Sf[(wave * 16 + j + hi8) * 64 + nt * 16 + cl] = accS[nt][j];
        __syncthreads();

        // ---- online softmax with softcap + causal mask (thread r = row) ----
        if (tid < 64) {
            const int r  = tid;
            const int qi = qb * 64 + r;
            float bm = -1e30f;
            for (int c = 0; c < 64; ++c) {
                float s = Sf[r * 64 + c] * 0.08838834764831845f;  // 1/sqrt(128)
                s = 50.0f * tanhf(s * 0.02f);                     // logit softcap
                if (kb * 64 + c > qi) s = -1e30f;                 // causal
                Sf[r * 64 + c] = s;
                bm = fmaxf(bm, s);
            }
            float newm = fmaxf(m_i, bm);
            float sc   = __expf(m_i - newm);
            float rs   = 0.f;
            for (int c = 0; c < 64; ++c) {
                float p = __expf(Sf[r * 64 + c] - newm);
                Sf[r * 64 + c] = p;
                rs += p;
            }
            l_i = l_i * sc + rs;
            m_i = newm;
            rowScale[r] = sc;
        }
        __syncthreads();

        // ---- rescale O, then O += P V ----
#pragma unroll
        for (int j = 0; j < 8; ++j) {
            float f = rowScale[wave * 16 + j + hi8];
#pragma unroll
            for (int vt = 0; vt < 8; ++vt) accO[vt][j] *= f;
        }
#pragma unroll
        for (int ks = 0; ks < 2; ++ks) {
            v16bf pf = frag_a_f32(Sf + ks * 32, wave * 16, 64, lane);
#pragma unroll
            for (int vt = 0; vt < 8; ++vt) {
                v16bf vf = frag_b(Vs + ks * 32, vt * 16, 64, lane);
                accO[vt] = wmma_bf16(pf, vf, accO[vt]);
            }
        }
        __syncthreads();   // Ks/Vs/Sf are overwritten next iteration
    }

    if (tid < 64) rowLinv[tid] = 1.0f / l_i;
    __syncthreads();

    // ---- epilogue: ctx bf16 [B,T,16,128] (matches out_kernel row order) ----
#pragma unroll
    for (int vt = 0; vt < 8; ++vt)
#pragma unroll
        for (int j = 0; j < 8; ++j) {
            int r  = wave * 16 + j + hi8;
            int qi = qb * 64 + r;
            float v = accO[vt][j] * rowLinv[r];
            Ctx[(((size_t)(b * SEQ + qi)) * NHEADS + hf) * HDIM + vt * 16 + cl] =
                f32_to_bf16(v);
        }
}

// ---------------------------------------------------------------------------
extern "C" void kernel_launch(void* const* d_in, const int* in_sizes, int n_in,
                              void* d_out, int out_size, void* d_ws, size_t ws_size,
                              hipStream_t stream) {
    (void)in_sizes; (void)n_in; (void)out_size; (void)ws_size;
    const float* x          = (const float*)d_in[0];
    /* d_in[1] = mask: causal, implicit */
    const float* q_kernel   = (const float*)d_in[2];
    const float* k_kernel   = (const float*)d_in[3];
    const float* v_kernel   = (const float*)d_in[4];
    const float* out_kernel = (const float*)d_in[5];
    float* out = (float*)d_out;

    const int M = BATCH * SEQ;                      // 4096
    const size_t MB = 1024 * 1024;
    char* ws = (char*)d_ws;
    float*          proj = (float*)ws;                         // 48 MB (f32 [M][3072])
    unsigned short* ctx  = (unsigned short*)ws;                // aliases proj (dead after rope)
    unsigned short* Qb   = (unsigned short*)(ws + 48  * MB);   // 16 MB
    unsigned short* Kb   = (unsigned short*)(ws + 64  * MB);   //  4 MB
    unsigned short* Vt   = (unsigned short*)(ws + 68  * MB);   //  4 MB
    unsigned short* xb   = (unsigned short*)(ws + 72  * MB);   // 16 MB  bf16 [M][2048]
    unsigned short* qkT  = (unsigned short*)(ws + 88  * MB);   //  8 MB  bf16 [2048][2048]
    unsigned short* kkT  = (unsigned short*)(ws + 96  * MB);   //  2 MB  bf16 [512][2048]
    unsigned short* vkT  = (unsigned short*)(ws + 98  * MB);   //  2 MB  bf16 [512][2048]
    unsigned short* okT  = (unsigned short*)(ws + 100 * MB);   //  8 MB  bf16 [2048][2048]

    // 0: one-time bf16 conversion (x) and transpose+convert (weights)
    cvt_f32_bf16<<<dim3((M * CDIM) / 1024), 256, 0, stream>>>(x, xb, M * CDIM);
    cvt_transpose_bf16<<<dim3(2048 / 32, 2048 / 32), 256, 0, stream>>>(q_kernel, qkT, CDIM, 2048);
    cvt_transpose_bf16<<<dim3(512 / 32,  2048 / 32), 256, 0, stream>>>(k_kernel, kkT, CDIM, 512);
    cvt_transpose_bf16<<<dim3(512 / 32,  2048 / 32), 256, 0, stream>>>(v_kernel, vkT, CDIM, 512);
    cvt_transpose_bf16<<<dim3(2048 / 32, 2048 / 32), 256, 0, stream>>>(out_kernel, okT, CDIM, 2048);

    // 1-3: QKV projections into proj [M x 3072] (TDM-staged WMMA GEMM)
    gemm_bf16_wmma<<<dim3(2048 / 128, M / 128), 128, 0, stream>>>(
        xb, qkT, proj, M, 2048, CDIM, QKV_N, 0);
    gemm_bf16_wmma<<<dim3(512 / 128, M / 128), 128, 0, stream>>>(
        xb, kkT, proj, M, 512, CDIM, QKV_N, 2048);
    gemm_bf16_wmma<<<dim3(512 / 128, M / 128), 128, 0, stream>>>(
        xb, vkT, proj, M, 512, CDIM, QKV_N, 2560);

    // 4: RoPE + bf16 repack (V transposed)
    rope_repack<<<dim3(SEQ, BATCH), 128, 0, stream>>>(proj, Qb, Kb, Vt);

    // 5: flash attention (writes ctx over the dead proj region)
    attention_kernel<<<dim3(SEQ / 64, NHEADS, BATCH), 128, 0, stream>>>(Qb, Kb, Vt, ctx);

    // 6: output projection -> d_out f32 [B,T,C]
    gemm_bf16_wmma<<<dim3(2048 / 128, M / 128), 128, 0, stream>>>(
        ctx, okT, out, M, CDIM, CDIM, CDIM, 0);
}
